// Noise_predictor_70884140253452
// MI455X (gfx1250) — matrix operand order repaired and checked by reading, not compiled, not run on hardware
//
#include <hip/hip_runtime.h>
#include <hip/hip_bf16.h>
#include <math.h>

// Problem constants (match reference)
#define Dm 512
#define Tt 20
#define Ff 66
#define Hh 8
#define Ll 8
#define Bb 512
#define BT (Bb * Tt)
#define DH 64   // Dm / Hh

// GEMM tiling
#define MB 128
#define NB 128
#define KB 32
#define KP 40   // padded K stride (in bf16 elements) for LDS tiles

typedef __attribute__((ext_vector_type(16))) __bf16 bf16x16;
typedef __attribute__((ext_vector_type(8)))  float  f32x8;

union FragAB { bf16x16 bf; uint4 q[2]; };
union FragC  { f32x8 v; float f[8]; };

__device__ __forceinline__ unsigned short f2bf(float f) {
  unsigned int u = __float_as_uint(f);
  u = (u + 0x7fffu + ((u >> 16) & 1u)) >> 16;   // round-to-nearest-even
  return (unsigned short)u;
}
__device__ __forceinline__ float gelu_f(float x) {
  return 0.5f * x * (1.0f + erff(x * 0.70710678118654752f)); // exact erf GELU
}
__device__ __forceinline__ float silu_f(float x) {
  return x / (1.0f + expf(-x));
}

// ---------------------------------------------------------------------------
// Generic WMMA GEMM: out(MxN) = epilogue(A(MxK) @ W + bias)
// transB==0: W is (K,N) row-major.  transB==1: W is (N,K) row-major.
// biasMode: 0 none, 1 bias[col], 2 bias[row % bmod]
// act: 0 none, 1 gelu, 2 silu
// storeMode: 0 out=v, 1 out+=v, 2 out=res-v, 3 out=res+v
// Block: 256 threads = 8 waves; block tile 128x128; wave tile 32x64
// (2 x 4 WMMA 16x16x32_bf16 accumulators per wave).
// ---------------------------------------------------------------------------
__global__ __launch_bounds__(256) void k_gemm(
    const float* __restrict__ A, const float* __restrict__ W,
    const float* __restrict__ bias, const float* __restrict__ res,
    float* __restrict__ out, int M, int N, int K,
    int transB, int biasMode, int bmod, int act, int storeMode)
{
  __shared__ __align__(16) unsigned short As[MB * KP];
  __shared__ __align__(16) unsigned short Bs[NB * KP];
  const int tid = threadIdx.x;
  const int bm = blockIdx.x * MB;
  const int bn = blockIdx.y * NB;
  const int lane = tid & 31;
  const int wid  = tid >> 5;
  const int wm = wid & 3;      // 4 wave-rows  (32 rows each)
  const int wn = wid >> 2;     // 2 wave-cols  (64 cols each)
  const int l15 = lane & 15;
  const int hi  = lane >> 4;

  FragC acc[2][4];
#pragma unroll
  for (int i = 0; i < 2; i++)
#pragma unroll
    for (int j = 0; j < 4; j++) acc[i][j].v = (f32x8)(0.0f);

  const bool a4 = ((K & 3) == 0);
  const bool w4 = transB ? ((K & 3) == 0) : ((N & 3) == 0);

  for (int k0 = 0; k0 < K; k0 += KB) {
    __syncthreads();
    // ---- stage A tile (MB x KB) fp32 -> bf16 ----
    {
      const int tr = tid >> 3;          // 0..31
      const int tc = (tid & 7) * 4;     // 0..28
#pragma unroll
      for (int rr = 0; rr < MB; rr += 32) {
        int row = bm + tr + rr;
        int kk = k0 + tc;
        float fx = 0.f, fy = 0.f, fz = 0.f, fw = 0.f;
        if (row < M) {
          if (a4 && kk + 3 < K) {
            float4 f = *(const float4*)(A + (size_t)row * K + kk);
            fx = f.x; fy = f.y; fz = f.z; fw = f.w;
          } else {
            const float* ap = A + (size_t)row * K;
            if (kk + 0 < K) fx = ap[kk + 0];
            if (kk + 1 < K) fy = ap[kk + 1];
            if (kk + 2 < K) fz = ap[kk + 2];
            if (kk + 3 < K) fw = ap[kk + 3];
          }
        }
        unsigned short* d = &As[(tr + rr) * KP + tc];
        d[0] = f2bf(fx); d[1] = f2bf(fy); d[2] = f2bf(fz); d[3] = f2bf(fw);
      }
    }
    // ---- stage B tile as Bs[n][k] (column-of-W contiguous in k) ----
    if (!transB) {
      const int tk = tid >> 5;          // 0..7
      const int tn = (tid & 31) * 4;    // 0..124
#pragma unroll
      for (int kk2 = 0; kk2 < KB; kk2 += 8) {
        int k = k0 + tk + kk2;
        int n = bn + tn;
        float fx = 0.f, fy = 0.f, fz = 0.f, fw = 0.f;
        if (k < K) {
          if (w4 && n + 3 < N) {
            float4 f = *(const float4*)(W + (size_t)k * N + n);
            fx = f.x; fy = f.y; fz = f.z; fw = f.w;
          } else {
            const float* wp = W + (size_t)k * N;
            if (n + 0 < N) fx = wp[n + 0];
            if (n + 1 < N) fy = wp[n + 1];
            if (n + 2 < N) fz = wp[n + 2];
            if (n + 3 < N) fw = wp[n + 3];
          }
        }
        int kk = tk + kk2;
        Bs[(tn + 0) * KP + kk] = f2bf(fx);
        Bs[(tn + 1) * KP + kk] = f2bf(fy);
        Bs[(tn + 2) * KP + kk] = f2bf(fz);
        Bs[(tn + 3) * KP + kk] = f2bf(fw);
      }
    } else {
      const int tr = tid >> 3;          // 0..31
      const int tc = (tid & 7) * 4;     // 0..28
#pragma unroll
      for (int rr = 0; rr < NB; rr += 32) {
        int n = bn + tr + rr;
        int kk = k0 + tc;
        float fx = 0.f, fy = 0.f, fz = 0.f, fw = 0.f;
        if (n < N) {
          if (w4 && kk + 3 < K) {
            float4 f = *(const float4*)(W + (size_t)n * K + kk);
            fx = f.x; fy = f.y; fz = f.z; fw = f.w;
          } else {
            const float* wp = W + (size_t)n * K;
            if (kk + 0 < K) fx = wp[kk + 0];
            if (kk + 1 < K) fy = wp[kk + 1];
            if (kk + 2 < K) fz = wp[kk + 2];
            if (kk + 3 < K) fw = wp[kk + 3];
          }
        }
        unsigned short* d = &Bs[(tr + rr) * KP + tc];
        d[0] = f2bf(fx); d[1] = f2bf(fy); d[2] = f2bf(fz); d[3] = f2bf(fw);
      }
    }
    __syncthreads();

    // ---- fragment loads per ISA layout ----
    // A 16x32 bf16: lane<16 -> K[0..7] in v0..3, K[16..23] in v4..7 (klo=0)
    //               lane>=16 -> K[8..15], K[24..31]                  (klo=8)
    FragAB afr[2], bfr[4];
    const int aklo = hi ? 8 : 0;
#pragma unroll
    for (int mt = 0; mt < 2; mt++) {
      int r = wm * 32 + mt * 16 + l15;
      afr[mt].q[0] = *(const uint4*)&As[r * KP + aklo];
      afr[mt].q[1] = *(const uint4*)&As[r * KP + aklo + 16];
    }
    // B 32x16 bf16: lane<16 -> col=lane, K[0..7], K[16..23]; lane>=16 -> K[8..15], K[24..31]
    const int bklo = hi ? 8 : 0;
#pragma unroll
    for (int nt = 0; nt < 4; nt++) {
      int c = wn * 64 + nt * 16 + l15;
      bfr[nt].q[0] = *(const uint4*)&Bs[c * KP + bklo];
      bfr[nt].q[1] = *(const uint4*)&Bs[c * KP + bklo + 16];
    }
#pragma unroll
    for (int mt = 0; mt < 2; mt++)
#pragma unroll
      for (int nt = 0; nt < 4; nt++)
        acc[mt][nt].v = __builtin_amdgcn_wmma_f32_16x16x32_bf16(
            false, afr[mt].bf, false, bfr[nt].bf,
            (short)0, acc[mt][nt].v, false, false);
  }

  // ---- epilogue: C VGPR r -> row = r + (lane>=16 ? 8 : 0), col = lane&15 ----
#pragma unroll
  for (int mt = 0; mt < 2; mt++) {
#pragma unroll
    for (int nt = 0; nt < 4; nt++) {
      int col = bn + wn * 64 + nt * 16 + l15;
      if (col >= N) continue;
#pragma unroll
      for (int r = 0; r < 8; r++) {
        int row = bm + wm * 32 + mt * 16 + hi * 8 + r;
        if (row >= M) continue;
        float v = acc[mt][nt].f[r];
        if (biasMode == 1)      v += bias[col];
        else if (biasMode == 2) v += bias[row % bmod];
        if (act == 1)      v = gelu_f(v);
        else if (act == 2) v = silu_f(v);
        size_t idx = (size_t)row * N + col;
        if (storeMode == 0)      out[idx] = v;
        else if (storeMode == 1) out[idx] += v;
        else if (storeMode == 2) out[idx] = res[idx] - v;
        else                     out[idx] = res[idx] + v;
      }
    }
  }
}

// ---------------------------------------------------------------------------
// LayerNorm over last dim (D=512); optional fused residual (x + resy)
// ---------------------------------------------------------------------------
__global__ __launch_bounds__(256) void k_ln(
    const float* __restrict__ x, const float* __restrict__ resy,
    const float* __restrict__ g, const float* __restrict__ b,
    float* __restrict__ out, int D)
{
  __shared__ float xs[Dm];
  __shared__ float red[256], red2[256];
  int row = blockIdx.x;
  int tid = threadIdx.x;
  const float* xr = x + (size_t)row * D;
  float s = 0.f, s2 = 0.f;
  for (int i = tid; i < D; i += 256) {
    float v = xr[i];
    if (resy) v += resy[(size_t)row * D + i];
    xs[i] = v; s += v; s2 += v * v;
  }
  red[tid] = s; red2[tid] = s2;
  __syncthreads();
  for (int off = 128; off > 0; off >>= 1) {
    if (tid < off) { red[tid] += red[tid + off]; red2[tid] += red2[tid + off]; }
    __syncthreads();
  }
  float mean = red[0] / (float)D;
  float var  = red2[0] / (float)D - mean * mean;
  float rs = rsqrtf(var + 1e-5f);
  for (int i = tid; i < D; i += 256)
    out[(size_t)row * D + i] = (xs[i] - mean) * rs * g[i] + b[i];
}

// ---------------------------------------------------------------------------
// tmix: out[b,n,d] = sum_m A[n,m] * X[b,m,d]   (A is TxT; transA reads A[m,n])
// optional per-d bias then gelu. grid = (D/64, B), block = 64
// ---------------------------------------------------------------------------
__global__ __launch_bounds__(64) void k_tmix(
    const float* __restrict__ Am, const float* __restrict__ X,
    const float* __restrict__ bias, float* __restrict__ out,
    int D, int transA, int act)
{
  __shared__ float xs[Tt][64];
  __shared__ float as[Tt * Tt];
  int b = blockIdx.y;
  int d0 = blockIdx.x * 64 + threadIdx.x;
  int tid = threadIdx.x;
#pragma unroll
  for (int m = 0; m < Tt; m++)
    xs[m][tid] = X[((size_t)(b * Tt + m)) * D + d0];
  for (int i = tid; i < Tt * Tt; i += 64) as[i] = Am[i];
  __syncthreads();
#pragma unroll
  for (int n = 0; n < Tt; n++) {
    float acc = 0.f;
#pragma unroll
    for (int m = 0; m < Tt; m++) {
      float a = transA ? as[m * Tt + n] : as[n * Tt + m];
      acc += a * xs[m][tid];
    }
    if (bias) acc += bias[d0];
    if (act == 1) acc = gelu_f(acc);
    out[((size_t)(b * Tt + n)) * D + d0] = acc;
  }
}

// ---------------------------------------------------------------------------
// Attention: per (b,h), 20x20 scores, softmax over keys, weighted sum of V.
// q/k/v/y are (B,T,D) with head layout D = H*64. grid = B*H, block = 64.
// ---------------------------------------------------------------------------
__global__ __launch_bounds__(64) void k_attn(
    const float* __restrict__ q, const float* __restrict__ k,
    const float* __restrict__ v, float* __restrict__ y)
{
  __shared__ float qs[Tt][DH], ks[Tt][DH], vs[Tt][DH];
  __shared__ float ss[Tt][Tt + 1];
  int b = blockIdx.x / Hh;
  int h = blockIdx.x % Hh;
  int tid = threadIdx.x;
  size_t base = (size_t)b * Tt * Dm + (size_t)h * DH;
#pragma unroll
  for (int n = 0; n < Tt; n++) {
    qs[n][tid] = q[base + (size_t)n * Dm + tid];
    ks[n][tid] = k[base + (size_t)n * Dm + tid];
    vs[n][tid] = v[base + (size_t)n * Dm + tid];
  }
  __syncthreads();
  const float scale = 0.125f; // 1/sqrt(64)
  for (int idx = tid; idx < Tt * Tt; idx += 64) {
    int n = idx / Tt, m = idx % Tt;
    float d = 0.f;
#pragma unroll
    for (int j = 0; j < DH; j++) d += qs[n][j] * ks[m][j];
    ss[n][m] = d * scale;
  }
  __syncthreads();
  if (tid < Tt) {
    int n = tid;
    float mx = -1e30f;
#pragma unroll
    for (int m = 0; m < Tt; m++) mx = fmaxf(mx, ss[n][m]);
    float sum = 0.f;
#pragma unroll
    for (int m = 0; m < Tt; m++) { float e = expf(ss[n][m] - mx); ss[n][m] = e; sum += e; }
    float inv = 1.f / sum;
#pragma unroll
    for (int m = 0; m < Tt; m++) ss[n][m] *= inv;
  }
  __syncthreads();
#pragma unroll
  for (int n = 0; n < Tt; n++) {
    float a = 0.f;
#pragma unroll
    for (int m = 0; m < Tt; m++) a += ss[n][m] * vs[m][tid];
    y[base + (size_t)n * Dm + tid] = a;
  }
}

// ---------------------------------------------------------------------------
// Timestep embedding: out[b, 0:256]=cos(t*f_i), out[b, 256:512]=sin(t*f_i)
// ---------------------------------------------------------------------------
__global__ __launch_bounds__(256) void k_tse(const int* __restrict__ t,
                                             float* __restrict__ out)
{
  int b = blockIdx.x;
  int i = threadIdx.x; // 0..255
  float tv = (float)t[b];
  float fr = expf(-logf(10000.0f) * (float)i / 256.0f);
  float arg = tv * fr;
  out[(size_t)b * Dm + i] = cosf(arg);
  out[(size_t)b * Dm + 256 + i] = sinf(arg);
}

// ---------------------------------------------------------------------------
// Elementwise adds
// ---------------------------------------------------------------------------
__global__ void k_add(const float* __restrict__ a, const float* __restrict__ c,
                      float* __restrict__ out, int n) {
  int i = blockIdx.x * 256 + threadIdx.x;
  if (i < n) out[i] = a[i] + c[i];
}
// out[i] = a[i] + c[i % P]   (seq_emb (T,D) broadcast over B)
__global__ void k_add_mod(const float* __restrict__ a, const float* __restrict__ c,
                          float* __restrict__ out, int n, int P) {
  int i = blockIdx.x * 256 + threadIdx.x;
  if (i < n) out[i] = a[i] + c[i % P];
}
// out[b,t,d] = a[b,t,d] + e[b,d]   (time emb (B,D) broadcast over T)
__global__ void k_add_mid(const float* __restrict__ a, const float* __restrict__ e,
                          float* __restrict__ out, int n) {
  int i = blockIdx.x * 256 + threadIdx.x;
  if (i < n) {
    int d = i % Dm;
    int b = (i / Dm) / Tt;
    out[i] = a[i] + e[(size_t)b * Dm + d];
  }
}

// ---------------------------------------------------------------------------
extern "C" void kernel_launch(void* const* d_in, const int* in_sizes, int n_in,
                              void* d_out, int out_size, void* d_ws, size_t ws_size,
                              hipStream_t stream) {
  (void)in_sizes; (void)n_in; (void)out_size; (void)ws_size;
  const float* x_in   = (const float*)d_in[0];
  const int*   tsteps = (const int*)d_in[1];
  const float* mod_in = (const float*)d_in[2];

  int pi = 3;
  auto Pp = [&]() { return (const float*)d_in[pi++]; };
  const float* joint_w = Pp(); const float* joint_b = Pp();
  // obse
  const float* lin1_w = Pp(); const float* lin1_b = Pp(); const float* seq_emb = Pp();
  const float* oatt[9]; for (int j = 0; j < 9; j++) oatt[j] = Pp();
  const float* offn[8]; for (int j = 0; j < 8; j++) offn[j] = Pp();
  // time
  const float* time_w1 = Pp(); const float* time_b1 = Pp();
  const float* time_w2 = Pp(); const float* time_b2 = Pp();
  // layers
  const float* Lffn1[Ll][8]; const float* Lca[Ll][9]; const float* Lffn2[Ll][8];
  const float* Ll3w[Ll]; const float* Ll3b[Ll]; const float* Ll4w[Ll]; const float* Ll4b[Ll];
  for (int l = 0; l < Ll; l++) {
    for (int j = 0; j < 8; j++) Lffn1[l][j] = Pp();
    for (int j = 0; j < 9; j++) Lca[l][j]  = Pp();
    for (int j = 0; j < 8; j++) Lffn2[l][j] = Pp();
    Ll3w[l] = Pp(); Ll3b[l] = Pp(); Ll4w[l] = Pp(); Ll4b[l] = Pp();
  }
  const float* out_w = Pp(); const float* out_b = Pp();

  // workspace layout
  float* ws = (float*)d_ws;
  const size_t nBTD = (size_t)BT * Dm;
  float* Eb = ws;                 // emb (B,T,D)
  float* Hb = Eb + nBTD;          // h
  float* AX = Hb + nBTD;          // all_x
  float* T0 = AX + nBTD;
  float* T1 = T0 + nBTD;
  float* T2 = T1 + nBTD;
  float* T3 = T2 + nBTD;
  float* T4 = T3 + nBTD;
  float* T5 = T4 + nBTD;
  float* S0 = T5 + nBTD;                  // (B,D)
  float* S1 = S0 + (size_t)Bb * Dm;
  float* S2 = S1 + (size_t)Bb * Dm;
  const int nb = (int)nBTD;

  auto gemm = [&](const float* A, const float* W, const float* bias,
                  const float* res, float* out, int M, int N, int K,
                  int transB, int biasMode, int bmod, int act, int storeMode) {
    dim3 g((M + MB - 1) / MB, (N + NB - 1) / NB);
    k_gemm<<<g, dim3(256), 0, stream>>>(A, W, bias, res, out, M, N, K,
                                        transB, biasMode, bmod, act, storeMode);
  };
  auto ln = [&](const float* xp, const float* yp, const float* g,
                const float* b, float* outp, int rows) {
    k_ln<<<dim3(rows), dim3(256), 0, stream>>>(xp, yp, g, b, outp, Dm);
  };
  auto tmix = [&](const float* Am, const float* X, const float* bias,
                  float* outp, int transA, int act) {
    k_tmix<<<dim3(Dm / 64, Bb), dim3(64), 0, stream>>>(Am, X, bias, outp, Dm, transA, act);
  };
  // fp: g1w, g1a, g1b, g2w, g2a, g2b, lw, lb
  auto ffn = [&](const float* xin, const float* const* fp, float* outp,
                 const float* resAdd, float* tA, float* tB) {
    gemm(xin, fp[0], nullptr, nullptr, tA, BT, Dm, Dm, 0, 0, 0, 0, 0);  // s = x@g1w
    tmix(fp[1], tA, fp[2], tB, 0, 1);                                   // gelu(g1a mix + g1b)
    tmix(fp[3], tB, nullptr, tA, 1, 0);                                 // z = g2w^T mix
    gemm(tA, fp[4], fp[5], nullptr, tB, BT, Dm, Dm, 1, 2, Tt, 1, 0);    // gelu(z@g2a^T + g2b[t])
    gemm(tB, fp[6], fp[7], resAdd, outp, BT, Dm, Dm, 0, 1, 0, 0, resAdd ? 3 : 0); // @lw + lb (+res)
  };
  // ap: n_g, n_b, tn_g, tn_b, wq, wk, wv, n2_g, n2_b
  auto attn = [&](const float* xp, const float* xfp, const float* const* ap, bool selfA,
                  float* outp, float* txn, float* txfn,
                  float* tq, float* tk, float* tv, float* ty) {
    ln(xp, nullptr, ap[0], ap[1], txn, BT);
    const float* xfn;
    if (selfA) { xfn = txn; }
    else { ln(xfp, nullptr, ap[2], ap[3], txfn, BT); xfn = txfn; }
    gemm(txn, ap[4], nullptr, nullptr, tq, BT, Dm, Dm, 0, 0, 0, 0, 0);
    gemm(xfn, ap[5], nullptr, nullptr, tk, BT, Dm, Dm, 0, 0, 0, 0, 0);
    gemm(xfn, ap[6], nullptr, nullptr, tv, BT, Dm, Dm, 0, 0, 0, 0, 0);
    k_attn<<<dim3(Bb * Hh), dim3(64), 0, stream>>>(tq, tk, tv, ty);
    ln(xp, ty, ap[7], ap[8], outp, BT);      // ln(x + y)
  };

  // ---- time embedding path: e2 (B,D) ----
  k_tse<<<dim3(Bb), dim3(256), 0, stream>>>(tsteps, S0);
  gemm(S0, time_w1, time_b1, nullptr, S1, Bb, Dm, Dm, 0, 1, 0, 2, 0);  // silu
  gemm(S1, time_w2, time_b2, nullptr, S2, Bb, Dm, Dm, 0, 1, 0, 0, 0);

  // ---- obse path ----
  gemm(mod_in, lin1_w, lin1_b, nullptr, T0, BT, Dm, Ff, 0, 1, 0, 1, 0);        // gelu
  k_add_mod<<<dim3((nb + 255) / 256), dim3(256), 0, stream>>>(T0, seq_emb, T0, nb, Tt * Dm);
  attn(T0, T0, oatt, true, T1, T2, T3, T3, T4, T5, T2);                        // m1 -> T1
  ffn(T1, offn, T2, T1, T3, T4);                                               // m2 = ffn(m1)+m1 -> T2
  k_add_mid<<<dim3((nb + 255) / 256), dim3(256), 0, stream>>>(T2, S2, Eb, nb); // E = m2 + e2

  // ---- h = x @ joint_w + joint_b ----
  gemm(x_in, joint_w, joint_b, nullptr, Hb, BT, Dm, Ff, 0, 1, 0, 0, 0);

  // ---- layers ----
  for (int l = 0; l < Ll; l++) {
    ffn(Hb, Lffn1[l], T2, nullptr, T0, T1);                      // x1 -> T2
    attn(Hb, Eb, Lca[l], false, T1, T0, T1, T3, T4, T5, T0);     // x2 -> T1
    ffn(T1, Lffn2[l], T5, T1, T0, T3);                           // x2f = ffn(x2)+x2 -> T5
    k_add<<<dim3((nb + 255) / 256), dim3(256), 0, stream>>>(T2, T5, T0, nb);  // x3 -> T0
    gemm(T0, Ll3w[l], Ll3b[l], Hb, Hb, BT, Dm, Dm, 0, 1, 0, 0, 2);            // h -= x3@l3w+l3b
    gemm(T0, Ll4w[l], Ll4b[l], AX, AX, BT, Dm, Dm, 0, 1, 0, 0, l == 0 ? 0 : 1); // all_x accumulate
  }

  // ---- output projection ----
  gemm(AX, out_w, out_b, nullptr, (float*)d_out, BT, Ff, Dm, 0, 1, 0, 0, 0);
}
